// GAT_48593259987027
// MI455X (gfx1250) — matrix operand order repaired and checked by reading, not compiled, not run on hardware
//
#include <hip/hip_runtime.h>
#include <stdint.h>

typedef __bf16 bf16_t;
typedef __attribute__((ext_vector_type(16))) __bf16 v16bf;
typedef __attribute__((ext_vector_type(8)))  __bf16 v8bf;
typedef __attribute__((ext_vector_type(8)))  float  v8f;

#define HEADS 16
#define HID   16
#define D1    256          // HEADS*HID
#define NFEAT 128
#define NCLS  10
#define NGR   64

// ---------- helpers ----------
__device__ __forceinline__ bf16_t f2bf(float f) {
  uint32_t u = __builtin_bit_cast(uint32_t, f);
  u += 0x7FFFu + ((u >> 16) & 1u);               // round-to-nearest-even
  uint16_t h = (uint16_t)(u >> 16);
  return __builtin_bit_cast(bf16_t, h);
}

__device__ __forceinline__ void atomicMaxF32(float* addr, float val) {
  // sign-aware monotone mapping: works for mixed-sign values, init = -inf
  if (val >= 0.0f) atomicMax((int*)addr, __float_as_int(val));
  else             atomicMin((unsigned int*)addr, __float_as_uint(val));
}

// ---------- init / convert ----------
__global__ void k_fill_f32(float* __restrict__ p, float v, int n) {
  int i = blockIdx.x * blockDim.x + threadIdx.x;
  if (i < n) p[i] = v;
}

__global__ void k_f32_to_bf16(const float* __restrict__ in, bf16_t* __restrict__ out, int n) {
  int i = blockIdx.x * blockDim.x + threadIdx.x;
  if (i < n) out[i] = f2bf(in[i]);
}

// W [K, Nout] f32 (row-major) -> Wt [Nout, K] bf16
__global__ void k_transpose_bf16(const float* __restrict__ W, bf16_t* __restrict__ Wt,
                                 int K, int Nout) {
  int i = blockIdx.x * blockDim.x + threadIdx.x;
  if (i >= K * Nout) return;
  int k = i / Nout, n = i - k * Nout;
  Wt[(size_t)n * K + k] = f2bf(W[i]);
}

// ---------- WMMA GEMM: C[M,Nout] = A[M,K](bf16) x Bt[Nout,K](bf16), f32 accum ----------
// A tile (16 x K) is shared by all 8 waves of the block -> async-copy it to LDS once,
// then each wave reads its fragments with ds_load; B streams from global (L2-resident).
template <int K>
__global__ void k_gemm_wmma(const bf16_t* __restrict__ A,
                            const bf16_t* __restrict__ Bt,
                            float* __restrict__ C, int M, int Nout) {
  __shared__ __align__(16) bf16_t sA[16 * K];

  const int tid  = threadIdx.x;
  const int lane = tid & 31;
  const int wave = tid >> 5;
  const int half = lane >> 4;      // which half-wave
  const int l16  = lane & 15;
  const int row0 = blockIdx.x * 16;
  const int col0 = (blockIdx.y * 8 + wave) * 16;

  // ---- async copy of A tile: 16*K*2 bytes in 16B chunks (ASYNCcnt-tracked) ----
  constexpr int CHUNKS = (16 * K * 2) / 16;      // 256 (K=128) or 512 (K=256)
  {
    const char* gbase = (const char*)(A + (size_t)row0 * K);
    for (int c = tid; c < CHUNKS; c += 256) {
      uint64_t gaddr = (uint64_t)(uintptr_t)(gbase + c * 16);
      uint32_t laddr = (uint32_t)(uintptr_t)((char*)sA + c * 16);  // flat low 32b == LDS offset
      asm volatile("global_load_async_to_lds_b128 %0, %1, off"
                   :: "v"(laddr), "v"(gaddr) : "memory");
    }
    asm volatile("s_wait_asynccnt 0x0" ::: "memory");  // drain this wave's async ops
  }
  __syncthreads();                                      // all waves' tiles landed

  if (row0 >= M || col0 >= Nout) return;                // uniform per wave (never taken here)

  // A fragment from LDS: lane half h holds row l16, K = k0+8h+j (j<8) / k0+16+8h+(j-8)
  const bf16_t* arow = sA + (size_t)l16 * K;
  // B fragment from global: lane half h holds col l16, K = k0+8h+j (j<8) / k0+16+8h+(j-8)
  const bf16_t* bcol = Bt + (size_t)(col0 + l16) * K;

  v8f acc = {};
#pragma unroll
  for (int k0 = 0; k0 < K; k0 += 32) {
    v8bf alo = *(const v8bf*)(arow + k0 + 8 * half);
    v8bf ahi = *(const v8bf*)(arow + k0 + 16 + 8 * half);
    v8bf blo = *(const v8bf*)(bcol + k0 + 8 * half);
    v8bf bhi = *(const v8bf*)(bcol + k0 + 16 + 8 * half);
    v16bf a, b;
#pragma unroll
    for (int j = 0; j < 8; ++j) {
      a[j] = alo[j]; a[j + 8] = ahi[j];
      b[j] = blo[j]; b[j + 8] = bhi[j];
    }
    acc = __builtin_amdgcn_wmma_f32_16x16x32_bf16(false, a, false, b,
                                                  (short)0, acc, false, false);
  }
  // C/D layout: VGPR r -> M = r + 8*half, N = l16
  float* crow = C + (size_t)(row0 + 8 * half) * Nout + col0 + l16;
#pragma unroll
  for (int r = 0; r < 8; ++r) crow[(size_t)r * Nout] = acc[r];
}

// ---------- attention coefficients: a_src/a_dst [N, HEADS] ----------
__global__ void k_att_coef(const float* __restrict__ h,
                           const float* __restrict__ att_s,
                           const float* __restrict__ att_d,
                           float* __restrict__ asrc, float* __restrict__ adst, int N) {
  int i = blockIdx.x * blockDim.x + threadIdx.x;   // over N*HEADS
  if (i >= N * HEADS) return;
  int n = i >> 4, hd = i & 15;
  const float* hp = h + (size_t)n * D1 + hd * HID;
  float ss = 0.f, sd = 0.f;
#pragma unroll
  for (int c = 0; c < HID; ++c) {
    float v = hp[c];
    ss += v * att_s[hd * HID + c];
    sd += v * att_d[hd * HID + c];
  }
  asrc[i] = ss;
  adst[i] = sd;
}

__device__ __forceinline__ void edge_sd(const int* ei, int e, int E, int& s, int& d) {
  if (e < E) { s = ei[e]; d = ei[E + e]; } else { s = e - E; d = e - E; }  // self-loops
}

// ---------- pass 1: segment max ----------
__global__ void k_edge_max(const int* __restrict__ ei,
                           const float* __restrict__ asrc, const float* __restrict__ adst,
                           float* __restrict__ mmax, int E, int N) {
  int e = blockIdx.x * blockDim.x + threadIdx.x;
  if (e >= E + N) return;
  int s, d; edge_sd(ei, e, E, s, d);
#pragma unroll
  for (int hd = 0; hd < HEADS; ++hd) {
    float v = asrc[s * HEADS + hd] + adst[d * HEADS + hd];
    v = v > 0.f ? v : 0.2f * v;                     // leaky_relu(0.2)
    atomicMaxF32(&mmax[d * HEADS + hd], v);
  }
}

// ---------- pass 2: exp-sum ----------
__global__ void k_edge_expsum(const int* __restrict__ ei,
                              const float* __restrict__ asrc, const float* __restrict__ adst,
                              const float* __restrict__ mmax, float* __restrict__ ssum,
                              int E, int N) {
  int e = blockIdx.x * blockDim.x + threadIdx.x;
  if (e >= E + N) return;
  int s, d; edge_sd(ei, e, E, s, d);
#pragma unroll
  for (int hd = 0; hd < HEADS; ++hd) {
    float v = asrc[s * HEADS + hd] + adst[d * HEADS + hd];
    v = v > 0.f ? v : 0.2f * v;
    float ex = __expf(v - mmax[d * HEADS + hd]);
    atomicAdd(&ssum[d * HEADS + hd], ex);
  }
}

// ---------- pass 3: weighted scatter-add; one thread per (edge, head) ----------
__global__ void k_edge_aggregate(const int* __restrict__ ei,
                                 const float* __restrict__ asrc, const float* __restrict__ adst,
                                 const float* __restrict__ mmax, const float* __restrict__ ssum,
                                 const float* __restrict__ h, float* __restrict__ agg,
                                 int E, int N) {
  int t = blockIdx.x * blockDim.x + threadIdx.x;   // (E+N)*HEADS = 13.6M < 2^31
  if (t >= (E + N) * HEADS) return;
  int e = t >> 4, hd = t & 15;
  int s, d; edge_sd(ei, e, E, s, d);
  float v = asrc[s * HEADS + hd] + adst[d * HEADS + hd];
  v = v > 0.f ? v : 0.2f * v;
  float alpha = __expf(v - mmax[d * HEADS + hd]) / (ssum[d * HEADS + hd] + 1e-16f);
  const float* hs = h + (size_t)s * D1 + hd * HID;
  float* ad = agg + (size_t)d * D1 + hd * HID;
#pragma unroll
  for (int c = 0; c < HID; ++c) atomicAdd(&ad[c], hs[c] * alpha);
}

// ---------- conv1 epilogue: +bias1, elu, bn1 -> bf16 for gemm2 ----------
__global__ void k_post1(const float* __restrict__ agg, const float* __restrict__ bias1,
                        const float* __restrict__ g, const float* __restrict__ b,
                        const float* __restrict__ mean, const float* __restrict__ var,
                        bf16_t* __restrict__ out, int N) {
  int i = blockIdx.x * blockDim.x + threadIdx.x;   // N*D1 = 12.8M
  if (i >= N * D1) return;
  int j = i & (D1 - 1);
  float v = agg[i] + bias1[j];
  v = v > 0.f ? v : (__expf(v) - 1.f);             // elu
  v = (v - mean[j]) * (g[j] * rsqrtf(var[j] + 1e-5f)) + b[j];
  out[i] = f2bf(v);
}

// ---------- conv2 epilogue: head-mean, +bias2, bn2, pooled scatter ----------
__global__ void k_post2(const float* __restrict__ agg, const float* __restrict__ bias2,
                        const float* __restrict__ g, const float* __restrict__ b,
                        const float* __restrict__ mean, const float* __restrict__ var,
                        const int* __restrict__ batch,
                        float* __restrict__ pooled, float* __restrict__ counts, int N) {
  int n = blockIdx.x * blockDim.x + threadIdx.x;
  if (n >= N) return;
  int gid = batch[n];
  atomicAdd(&counts[gid], 1.0f);
  const float* row = agg + (size_t)n * D1;
#pragma unroll
  for (int c = 0; c < HID; ++c) {
    float acc = 0.f;
#pragma unroll
    for (int hd = 0; hd < HEADS; ++hd) acc += row[hd * HID + c];
    float v = acc * (1.0f / HEADS) + bias2[c];
    v = (v - mean[c]) * (g[c] * rsqrtf(var[c] + 1e-5f)) + b[c];
    atomicAdd(&pooled[gid * HID + c], v);
  }
}

// ---------- final: (pooled/counts) @ lin_W + lin_b ----------
__global__ void k_final(const float* __restrict__ pooled, const float* __restrict__ counts,
                        const float* __restrict__ lw, const float* __restrict__ lb,
                        float* __restrict__ out) {
  int i = blockIdx.x * blockDim.x + threadIdx.x;   // NGR*NCLS = 640
  if (i >= NGR * NCLS) return;
  int g = i / NCLS, k = i - g * NCLS;
  float cnt = counts[g]; cnt = cnt > 1.f ? cnt : 1.f;
  float acc = lb[k];
#pragma unroll
  for (int c = 0; c < HID; ++c) acc += (pooled[g * HID + c] / cnt) * lw[c * NCLS + k];
  out[i] = acc;
}

// =====================================================================
extern "C" void kernel_launch(void* const* d_in, const int* in_sizes, int n_in,
                              void* d_out, int out_size, void* d_ws, size_t ws_size,
                              hipStream_t stream) {
  const float* x        = (const float*)d_in[0];
  const int*   ei       = (const int*)  d_in[1];
  const int*   batch    = (const int*)  d_in[2];
  const float* W1       = (const float*)d_in[3];
  const float* att_s1   = (const float*)d_in[4];
  const float* att_d1   = (const float*)d_in[5];
  const float* bias1    = (const float*)d_in[6];
  const float* bn1g     = (const float*)d_in[7];
  const float* bn1b     = (const float*)d_in[8];
  const float* bn1m     = (const float*)d_in[9];
  const float* bn1v     = (const float*)d_in[10];
  const float* W2       = (const float*)d_in[11];
  const float* att_s2   = (const float*)d_in[12];
  const float* att_d2   = (const float*)d_in[13];
  const float* bias2    = (const float*)d_in[14];
  const float* bn2g     = (const float*)d_in[15];
  const float* bn2b     = (const float*)d_in[16];
  const float* bn2m     = (const float*)d_in[17];
  const float* bn2v     = (const float*)d_in[18];
  const float* linW     = (const float*)d_in[19];
  const float* linb     = (const float*)d_in[20];
  float* out = (float*)d_out;

  const int N  = in_sizes[0] / NFEAT;   // 50000
  const int E  = in_sizes[1] / 2;       // 800000
  const int ET = E + N;                 // with self-loops

  // ---- carve workspace (256B aligned) ----
  char* wp = (char*)d_ws;
  auto carve = [&](size_t bytes) -> void* {
    void* p = wp; wp += (bytes + 255) & ~(size_t)255; return p;
  };
  bf16_t* xbf   = (bf16_t*)carve((size_t)N * NFEAT * 2);
  bf16_t* w1t   = (bf16_t*)carve((size_t)D1 * NFEAT * 2);
  bf16_t* w2t   = (bf16_t*)carve((size_t)D1 * D1 * 2);
  float*  h     = (float*) carve((size_t)N * D1 * 4);
  float*  agg   = (float*) carve((size_t)N * D1 * 4);
  bf16_t* hbf   = (bf16_t*)carve((size_t)N * D1 * 2);
  float*  asrc  = (float*) carve((size_t)N * HEADS * 4);
  float*  adst  = (float*) carve((size_t)N * HEADS * 4);
  float*  mmax  = (float*) carve((size_t)N * HEADS * 4);
  float*  ssum  = (float*) carve((size_t)N * HEADS * 4);
  float*  pooled= (float*) carve((size_t)NGR * HID * 4);
  float*  counts= (float*) carve((size_t)NGR * 4);

  const int B = 256;
  auto nb = [](int n, int b) { return (n + b - 1) / b; };
  const float NEG_INF = -3.4e38f;

  // ---- precision conversion / transposes ----
  k_f32_to_bf16<<<nb(N * NFEAT, B), B, 0, stream>>>(x, xbf, N * NFEAT);
  k_transpose_bf16<<<nb(NFEAT * D1, B), B, 0, stream>>>(W1, w1t, NFEAT, D1);
  k_transpose_bf16<<<nb(D1 * D1, B), B, 0, stream>>>(W2, w2t, D1, D1);

  // ================= conv1 =================
  k_fill_f32<<<nb(N * D1, B), B, 0, stream>>>(agg, 0.f, N * D1);
  k_fill_f32<<<nb(N * HEADS, B), B, 0, stream>>>(mmax, NEG_INF, N * HEADS);
  k_fill_f32<<<nb(N * HEADS, B), B, 0, stream>>>(ssum, 0.f, N * HEADS);

  // h = x @ W1  (bf16 WMMA, f32 accum; A tile staged in LDS via async copy)
  k_gemm_wmma<NFEAT><<<dim3(N / 16, D1 / (16 * 8)), B, 0, stream>>>(xbf, w1t, h, N, D1);

  k_att_coef<<<nb(N * HEADS, B), B, 0, stream>>>(h, att_s1, att_d1, asrc, adst, N);
  k_edge_max<<<nb(ET, B), B, 0, stream>>>(ei, asrc, adst, mmax, E, N);
  k_edge_expsum<<<nb(ET, B), B, 0, stream>>>(ei, asrc, adst, mmax, ssum, E, N);
  k_edge_aggregate<<<nb(ET * HEADS, B), B, 0, stream>>>(ei, asrc, adst, mmax, ssum, h, agg, E, N);

  k_post1<<<nb(N * D1, B), B, 0, stream>>>(agg, bias1, bn1g, bn1b, bn1m, bn1v, hbf, N);

  // ================= conv2 =================
  k_fill_f32<<<nb(N * D1, B), B, 0, stream>>>(agg, 0.f, N * D1);
  k_fill_f32<<<nb(N * HEADS, B), B, 0, stream>>>(mmax, NEG_INF, N * HEADS);
  k_fill_f32<<<nb(N * HEADS, B), B, 0, stream>>>(ssum, 0.f, N * HEADS);

  // h = bn1(elu(conv1)) @ W2
  k_gemm_wmma<D1><<<dim3(N / 16, D1 / (16 * 8)), B, 0, stream>>>(hbf, w2t, h, N, D1);

  k_att_coef<<<nb(N * HEADS, B), B, 0, stream>>>(h, att_s2, att_d2, asrc, adst, N);
  k_edge_max<<<nb(ET, B), B, 0, stream>>>(ei, asrc, adst, mmax, E, N);
  k_edge_expsum<<<nb(ET, B), B, 0, stream>>>(ei, asrc, adst, mmax, ssum, E, N);
  k_edge_aggregate<<<nb(ET * HEADS, B), B, 0, stream>>>(ei, asrc, adst, mmax, ssum, h, agg, E, N);

  // ================= pool + classifier =================
  k_fill_f32<<<nb(NGR * HID, B), B, 0, stream>>>(pooled, 0.f, NGR * HID);
  k_fill_f32<<<1, B, 0, stream>>>(counts, 0.f, NGR);
  k_post2<<<nb(N, B), B, 0, stream>>>(agg, bias2, bn2g, bn2b, bn2m, bn2v, batch,
                                      pooled, counts, N);
  k_final<<<nb(NGR * NCLS, B), B, 0, stream>>>(pooled, counts, linW, linb, out);
}
